// LongFormer_57363583205852
// MI455X (gfx1250) — matrix operand order repaired and checked by reading, not compiled, not run on hardware
//
#include <hip/hip_runtime.h>
#include <hip/hip_bf16.h>

typedef __attribute__((ext_vector_type(16))) _Float16 v16h;
typedef __attribute__((ext_vector_type(2)))  _Float16 v2h;
typedef __attribute__((ext_vector_type(8)))  float    v8f;

#define S_LEN   4096
#define D_MODEL 512
#define NHEAD   8
#define DHEAD   64
#define NLAYER  12
#define FFN     1024
#define WIN     256
#define HEADV   339
#define CONCAT  1344

// ---------------------------------------------------------------------------
// WMMA fragment loaders (contiguous k order, wave32)
// A 16x32 f16: lane<16 -> row=lane,  elems 0..15 = K 0..15
//              lane>=16-> row=lane-16, elems 0..15 = K 16..31
// ---------------------------------------------------------------------------
__device__ __forceinline__ v16h load_a_frag(const _Float16* lds, int row,
                                            int kbase, int stride, int lane) {
  int r = row + (lane & 15);
  int khalf = (lane >> 4) * 16;
  const _Float16* p = lds + (size_t)r * stride + kbase + khalf;
  v16h out;
#pragma unroll
  for (int v = 0; v < 4; ++v) { out[2*v] = p[2*v]; out[2*v+1] = p[2*v+1]; }
#pragma unroll
  for (int v = 0; v < 4; ++v) { out[8+2*v] = p[8+2*v]; out[8+2*v+1] = p[8+2*v+1]; }
  return out;
}

// B 32x16 f16 from a transposed [N][K] LDS buffer: lane<16 -> col=lane, K 0..15
// lane>=16 -> col=lane-16, K 16..31 ; 16 contiguous f16 per lane.
__device__ __forceinline__ v16h load_b_frag(const _Float16* ldsT, int col,
                                            int kbase, int stride, int lane) {
  int n = col + (lane & 15);
  int kh = (lane >> 4) * 16;
  const _Float16* p = ldsT + (size_t)n * stride + kbase + kh;
  v16h out;
#pragma unroll
  for (int i = 0; i < 16; ++i) out[i] = p[i];
  return out;
}

// ---------------------------------------------------------------------------
// 1) Gather scaled embeddings into concat matrix [S, 1344]
// ---------------------------------------------------------------------------
__global__ void gather_kernel(const int* __restrict__ data,
                              const float* e0, const float* e1, const float* e2,
                              const float* e3, const float* e4, const float* e5,
                              float* __restrict__ concat) {
  int idx = blockIdx.x * blockDim.x + threadIdx.x;
  if (idx >= S_LEN * CONCAT) return;
  int t = idx / CONCAT, c = idx % CONCAT;
  int f, off, emb; float sc; const float* tab;
  if (c < 128)       { f = 0; off = 0;    emb = 128; sc = 11.313708499f; tab = e0; }
  else if (c < 384)  { f = 1; off = 128;  emb = 256; sc = 16.0f;         tab = e1; }
  else if (c < 448)  { f = 2; off = 384;  emb = 64;  sc = 8.0f;          tab = e2; }
  else if (c < 960)  { f = 3; off = 448;  emb = 512; sc = 22.627416998f; tab = e3; }
  else if (c < 1216) { f = 4; off = 960;  emb = 256; sc = 16.0f;         tab = e4; }
  else               { f = 5; off = 1216; emb = 128; sc = 11.313708499f; tab = e5; }
  int tok = data[t * 6 + f];
  concat[idx] = tab[(size_t)tok * emb + (c - off)] * sc;
}

// ---------------------------------------------------------------------------
// 2) WMMA GEMM: C[M,N] = epilogue(A[M,K] * B[K,N] + bias[N])
//    mode 0: none ; mode 1: *0.125 (Q scale) ; mode 2: exact GELU
//    Block 256 thr (8 waves). Tile 128x64, K step 32. Wave -> 32x32.
// ---------------------------------------------------------------------------
__global__ __launch_bounds__(256)
void gemm_kernel(const float* __restrict__ A, const float* __restrict__ B,
                 const float* __restrict__ bias, float* __restrict__ C,
                 int M, int N, int K, int mode) {
  const int n0 = blockIdx.x * 64;
  const int m0 = blockIdx.y * 128;
  const int tid = threadIdx.x, lane = tid & 31, wave = tid >> 5;
  const int wr = (wave >> 1) * 32;   // wave row offset 0/32/64/96
  const int wc = (wave & 1) * 32;    // wave col offset 0/32

  __shared__ _Float16 sa[128 * 32];   // A tile, row-major [128][32]
  __shared__ _Float16 sbt[64 * 32];   // B tile transposed [N=64][K=32]

  const bool full_n = (n0 + 64 <= N);

  // thread -> A half-row (16 f32)
  const int a_r = tid >> 1, a_c = (tid & 1) * 16;
  // fast-path thread -> B 2(k) x 4(n) subtile
  const int b_k = (tid >> 4) * 2;        // 0,2,..,30
  const int b_n = (tid & 15) * 4;        // 0,4,..,60

  v8f zero = {0.f,0.f,0.f,0.f,0.f,0.f,0.f,0.f};
  v8f acc[2][2];
  acc[0][0] = zero; acc[0][1] = zero; acc[1][0] = zero; acc[1][1] = zero;

  for (int k0 = 0; k0 < K; k0 += 32) {
    { // ---- A tile: 4x float4, convert, 2x ds_store_b128
      const float4* ap = (const float4*)(A + (size_t)(m0 + a_r) * K + k0 + a_c);
      float4 q0 = ap[0], q1 = ap[1], q2 = ap[2], q3 = ap[3];
      _Float16* sd = sa + a_r * 32 + a_c;
      sd[0]=(_Float16)q0.x;  sd[1]=(_Float16)q0.y;  sd[2]=(_Float16)q0.z;  sd[3]=(_Float16)q0.w;
      sd[4]=(_Float16)q1.x;  sd[5]=(_Float16)q1.y;  sd[6]=(_Float16)q1.z;  sd[7]=(_Float16)q1.w;
      sd[8]=(_Float16)q2.x;  sd[9]=(_Float16)q2.y;  sd[10]=(_Float16)q2.z; sd[11]=(_Float16)q2.w;
      sd[12]=(_Float16)q3.x; sd[13]=(_Float16)q3.y; sd[14]=(_Float16)q3.z; sd[15]=(_Float16)q3.w;
    }
    if (full_n) { // ---- B tile fast path: 2x float4 loads, 4x ds_store_b32
      const float* brow0 = B + (size_t)(k0 + b_k)     * N + n0 + b_n;
      const float* brow1 = B + (size_t)(k0 + b_k + 1) * N + n0 + b_n;
      float4 r0 = *(const float4*)brow0;
      float4 r1 = *(const float4*)brow1;
      float r0a[4] = {r0.x, r0.y, r0.z, r0.w};
      float r1a[4] = {r1.x, r1.y, r1.z, r1.w};
#pragma unroll
      for (int j = 0; j < 4; ++j) {
        v2h pp; pp[0] = (_Float16)r0a[j]; pp[1] = (_Float16)r1a[j];
        *(v2h*)(sbt + (b_n + j) * 32 + b_k) = pp;
      }
    } else {      // ---- B tile tail path (last N-tile of the 339-proj only)
      int c  = tid >> 3;
      int nb = (tid & 7) * 8;
      const float* bp = B + (size_t)(k0 + c) * N + n0 + nb;
#pragma unroll
      for (int j = 0; j < 8; ++j) {
        float v = (n0 + nb + j < N) ? bp[j] : 0.f;
        sbt[(nb + j) * 32 + c] = (_Float16)v;
      }
    }
    // prefetch next K-step tiles into L2/L0 while this step computes
    if (k0 + 32 < K) {
      __builtin_prefetch(A + (size_t)(m0 + a_r) * K + k0 + 32 + a_c, 0, 1);
      if (full_n)
        __builtin_prefetch(B + (size_t)(k0 + 32 + b_k) * N + n0 + b_n, 0, 1);
    }
    __syncthreads();

    v16h a0 = load_a_frag(sa, wr,      0, 32, lane);
    v16h a1 = load_a_frag(sa, wr + 16, 0, 32, lane);
    v16h b0 = load_b_frag(sbt, wc,      0, 32, lane);
    v16h b1 = load_b_frag(sbt, wc + 16, 0, 32, lane);
    acc[0][0] = __builtin_amdgcn_wmma_f32_16x16x32_f16(false, a0, false, b0, (short)0, acc[0][0], false, false);
    acc[0][1] = __builtin_amdgcn_wmma_f32_16x16x32_f16(false, a0, false, b1, (short)0, acc[0][1], false, false);
    acc[1][0] = __builtin_amdgcn_wmma_f32_16x16x32_f16(false, a1, false, b0, (short)0, acc[1][0], false, false);
    acc[1][1] = __builtin_amdgcn_wmma_f32_16x16x32_f16(false, a1, false, b1, (short)0, acc[1][1], false, false);
    __syncthreads();
  }

#pragma unroll
  for (int tr = 0; tr < 2; ++tr)
#pragma unroll
    for (int tc = 0; tc < 2; ++tc) {
      int col = n0 + wc + tc * 16 + (lane & 15);
      if (col < N) {
        float bv = bias[col];
#pragma unroll
        for (int i = 0; i < 8; ++i) {
          int row = m0 + wr + tr * 16 + i + ((lane >> 4) << 3);
          float v = acc[tr][tc][i] + bv;
          if (mode == 1)      v *= 0.125f;                               // 1/sqrt(64)
          else if (mode == 2) v = 0.5f * v * (1.f + erff(v * 0.70710678118f));
          C[(size_t)row * N + col] = v;
        }
      }
    }
}

// ---------------------------------------------------------------------------
// 3) Fused (residual) add + LayerNorm over D=512.  One block per row.
//    out = LN(src + addRow? + addBcast?) * g + b
// ---------------------------------------------------------------------------
__global__ __launch_bounds__(256)
void ln_kernel(const float* __restrict__ src, const float* __restrict__ addRow,
               const float* __restrict__ addB, const float* __restrict__ g,
               const float* __restrict__ b, float* __restrict__ out) {
  int row = blockIdx.x, tid = threadIdx.x;
  __shared__ float red[256];
  size_t base = (size_t)row * D_MODEL;
  float x0 = src[base + tid], x1 = src[base + 256 + tid];
  if (addRow) { x0 += addRow[base + tid]; x1 += addRow[base + 256 + tid]; }
  if (addB)   { x0 += addB[tid];          x1 += addB[256 + tid]; }

  red[tid] = x0 + x1; __syncthreads();
  for (int s = 128; s > 0; s >>= 1) { if (tid < s) red[tid] += red[tid + s]; __syncthreads(); }
  float mean = red[0] * (1.f / 512.f); __syncthreads();

  float d0 = x0 - mean, d1 = x1 - mean;
  red[tid] = d0 * d0 + d1 * d1; __syncthreads();
  for (int s = 128; s > 0; s >>= 1) { if (tid < s) red[tid] += red[tid + s]; __syncthreads(); }
  float rstd = rsqrtf(red[0] * (1.f / 512.f) + 1e-12f);

  out[base + tid]       = d0 * rstd * g[tid]       + b[tid];
  out[base + 256 + tid] = d1 * rstd * g[256 + tid] + b[256 + tid];
}

// ---------------------------------------------------------------------------
// 4) Sliding-window flash attention (window +/-256), WMMA QK^T and PV.
//    Block = 128 threads (4 waves) = one 64-query strip of one head.
//    Each wave owns 16 queries; 9 key blocks of 64 keys stream through LDS.
// ---------------------------------------------------------------------------
__global__ __launch_bounds__(128)
void attn_kernel(const float* __restrict__ Q, const float* __restrict__ K,
                 const float* __restrict__ V, const float* __restrict__ masks,
                 float* __restrict__ O) {
  const int strip = blockIdx.x;
  const int h     = blockIdx.y;
  const int tid = threadIdx.x, lane = tid & 31, wave = tid >> 5;
  const int q0_block = strip * 64;
  const int q0_wave  = q0_block + wave * 16;

  __shared__ _Float16 lds_q[64 * 64];    // [q][dh]
  __shared__ _Float16 lds_k[64 * 64];    // [key][dh]  (== B^T for QK^T)
  __shared__ _Float16 lds_vt[64 * 64];   // [dh][key]  (== B^T for PV)
  __shared__ _Float16 lds_p[4 * 16 * 64];// per-wave P stage [16 q][64 key]
  __shared__ float    kv_ok[64];

  // Q strip load: float4-vectorized, always in-bounds
  for (int i4 = tid; i4 < 64 * 16; i4 += 128) {
    int r = i4 >> 4, c4 = (i4 & 15) * 4;
    float4 qv = *(const float4*)(Q + (size_t)(q0_block + r) * D_MODEL + h * DHEAD + c4);
    _Float16* qd = lds_q + r * 64 + c4;
    qd[0] = (_Float16)qv.x; qd[1] = (_Float16)qv.y;
    qd[2] = (_Float16)qv.z; qd[3] = (_Float16)qv.w;
  }
  __syncthreads();
  v16h qa0 = load_a_frag(lds_q, wave * 16, 0,  64, lane);
  v16h qa1 = load_a_frag(lds_q, wave * 16, 32, 64, lane);

  v8f zero = {0.f,0.f,0.f,0.f,0.f,0.f,0.f,0.f};
  float m_run[8], l_run[8];
  v8f acc[4];
#pragma unroll
  for (int i = 0; i < 8; ++i) { m_run[i] = -1e30f; l_run[i] = 0.f; }
#pragma unroll
  for (int d = 0; d < 4; ++d) acc[d] = zero;

  const int kstart = q0_block - WIN;
  for (int bblk = 0; bblk < 9; ++bblk) {
    int kb = kstart + bblk * 64;
    // K/V tile load: float4-vectorized with row bounds guard
    for (int i4 = tid; i4 < 64 * 16; i4 += 128) {
      int r = i4 >> 4, c4 = (i4 & 15) * 4;
      int ka = kb + r;
      float4 kvv = {0.f,0.f,0.f,0.f}, vvv = {0.f,0.f,0.f,0.f};
      if (ka >= 0 && ka < S_LEN) {
        kvv = *(const float4*)(K + (size_t)ka * D_MODEL + h * DHEAD + c4);
        vvv = *(const float4*)(V + (size_t)ka * D_MODEL + h * DHEAD + c4);
      }
      _Float16* kd = lds_k + r * 64 + c4;
      kd[0] = (_Float16)kvv.x; kd[1] = (_Float16)kvv.y;
      kd[2] = (_Float16)kvv.z; kd[3] = (_Float16)kvv.w;
      lds_vt[(c4 + 0) * 64 + r] = (_Float16)vvv.x;
      lds_vt[(c4 + 1) * 64 + r] = (_Float16)vvv.y;
      lds_vt[(c4 + 2) * 64 + r] = (_Float16)vvv.z;
      lds_vt[(c4 + 3) * 64 + r] = (_Float16)vvv.w;
    }
    if (tid < 64) {
      int ka = kb + tid;
      kv_ok[tid] = (ka >= 0 && ka < S_LEN && masks[ka] > 0.f) ? 1.f : 0.f;
    }
    __syncthreads();

    // ---- scores: 4 tiles of 16 keys, K-depth = DH = 64 (2 WMMA each)
    v8f st[4];
#pragma unroll
    for (int t = 0; t < 4; ++t) {
      v8f c0 = zero;
      v16h kb0 = load_b_frag(lds_k, t * 16, 0,  64, lane);
      c0 = __builtin_amdgcn_wmma_f32_16x16x32_f16(false, qa0, false, kb0, (short)0, c0, false, false);
      v16h kb1 = load_b_frag(lds_k, t * 16, 32, 64, lane);
      c0 = __builtin_amdgcn_wmma_f32_16x16x32_f16(false, qa1, false, kb1, (short)0, c0, false, false);
      st[t] = c0;
    }

    // ---- band + bounds mask, block row-max
    float bmax[8];
#pragma unroll
    for (int i = 0; i < 8; ++i) bmax[i] = -1e30f;
#pragma unroll
    for (int t = 0; t < 4; ++t) {
      int kcol = kb + t * 16 + (lane & 15);
      float ok_k = kv_ok[t * 16 + (lane & 15)];
#pragma unroll
      for (int i = 0; i < 8; ++i) {
        int qr = q0_wave + i + ((lane >> 4) << 3);
        bool ok = (ok_k > 0.f) && (kcol >= qr - WIN) && (kcol <= qr + WIN);
        float sv = ok ? st[t][i] : -1e30f;
        st[t][i] = sv;
        bmax[i] = fmaxf(bmax[i], sv);
      }
    }
#pragma unroll
    for (int x = 1; x < 16; x <<= 1)
#pragma unroll
      for (int i = 0; i < 8; ++i) bmax[i] = fmaxf(bmax[i], __shfl_xor(bmax[i], x, 32));

    // ---- online softmax
    float scalef[8], rsum[8];
#pragma unroll
    for (int i = 0; i < 8; ++i) {
      float mn = fmaxf(m_run[i], bmax[i]);
      scalef[i] = expf(m_run[i] - mn);
      m_run[i] = mn;
      rsum[i] = 0.f;
    }
#pragma unroll
    for (int t = 0; t < 4; ++t)
#pragma unroll
      for (int i = 0; i < 8; ++i) {
        float p = expf(st[t][i] - m_run[i]);
        st[t][i] = p;
        rsum[i] += p;
      }
#pragma unroll
    for (int x = 1; x < 16; x <<= 1)
#pragma unroll
      for (int i = 0; i < 8; ++i) rsum[i] += __shfl_xor(rsum[i], x, 32);
#pragma unroll
    for (int i = 0; i < 8; ++i) l_run[i] = l_run[i] * scalef[i] + rsum[i];
#pragma unroll
    for (int d = 0; d < 4; ++d)
#pragma unroll
      for (int i = 0; i < 8; ++i) acc[d][i] *= scalef[i];

    // ---- stage P (f16) into per-wave LDS, re-read as A-fragments
    _Float16* pst = lds_p + wave * (16 * 64);
#pragma unroll
    for (int t = 0; t < 4; ++t)
#pragma unroll
      for (int i = 0; i < 8; ++i) {
        int prow = i + ((lane >> 4) << 3);
        pst[prow * 64 + t * 16 + (lane & 15)] = (_Float16)st[t][i];
      }
    asm volatile("s_wait_dscnt 0" ::: "memory");
    v16h pa0 = load_a_frag(pst, 0, 0,  64, lane);
    v16h pa1 = load_a_frag(pst, 0, 32, 64, lane);

    // ---- PV: out[16q x 64dh] += P[16x64] * V[64x64]
#pragma unroll
    for (int d = 0; d < 4; ++d) {
      v16h vb0 = load_b_frag(lds_vt, d * 16, 0,  64, lane);
      acc[d] = __builtin_amdgcn_wmma_f32_16x16x32_f16(false, pa0, false, vb0, (short)0, acc[d], false, false);
      v16h vb1 = load_b_frag(lds_vt, d * 16, 32, 64, lane);
      acc[d] = __builtin_amdgcn_wmma_f32_16x16x32_f16(false, pa1, false, vb1, (short)0, acc[d], false, false);
    }
    __syncthreads();
  }

#pragma unroll
  for (int d = 0; d < 4; ++d)
#pragma unroll
    for (int i = 0; i < 8; ++i) {
      int row = q0_wave + i + ((lane >> 4) << 3);
      int dh  = d * 16 + (lane & 15);
      float denom = l_run[i];
      O[(size_t)row * D_MODEL + h * DHEAD + dh] = acc[d][i] / (denom > 0.f ? denom : 1.f);
    }
}

// ---------------------------------------------------------------------------
// Host orchestration
// ---------------------------------------------------------------------------
extern "C" void kernel_launch(void* const* d_in, const int* in_sizes, int n_in,
                              void* d_out, int out_size, void* d_ws, size_t ws_size,
                              hipStream_t stream) {
  const int*   data   = (const int*)  d_in[0];
  const float* masks  = (const float*)d_in[1];
  const float* e0     = (const float*)d_in[2];
  const float* e1     = (const float*)d_in[3];
  const float* e2     = (const float*)d_in[4];
  const float* e3     = (const float*)d_in[5];
  const float* e4     = (const float*)d_in[6];
  const float* e5     = (const float*)d_in[7];
  const float* in_W   = (const float*)d_in[8];
  const float* in_b   = (const float*)d_in[9];
  const float* pos    = (const float*)d_in[10];
  const float* tok    = (const float*)d_in[11];
  const float* eg     = (const float*)d_in[12];
  const float* eb     = (const float*)d_in[13];
  const float* Wq     = (const float*)d_in[14];
  const float* bq     = (const float*)d_in[15];
  const float* Wk     = (const float*)d_in[16];
  const float* bk     = (const float*)d_in[17];
  const float* Wv     = (const float*)d_in[18];
  const float* bv     = (const float*)d_in[19];
  const float* Wo     = (const float*)d_in[20];
  const float* bo     = (const float*)d_in[21];
  const float* ln1g   = (const float*)d_in[22];
  const float* ln1b   = (const float*)d_in[23];
  const float* Wf1    = (const float*)d_in[24];
  const float* bf1    = (const float*)d_in[25];
  const float* Wf2    = (const float*)d_in[26];
  const float* bf2    = (const float*)d_in[27];
  const float* ln2g   = (const float*)d_in[28];
  const float* ln2b   = (const float*)d_in[29];
  const float* projW  = (const float*)d_in[30];
  const float* projb  = (const float*)d_in[31];
  float* out = (float*)d_out;

  float* ws = (float*)d_ws;
  const size_t SD = (size_t)S_LEN * D_MODEL;       // 2,097,152
  float* x  = ws;                                  // [S,D]
  float* q  = ws + 1 * SD;
  float* k  = ws + 2 * SD;
  float* v  = ws + 3 * SD;
  float* a  = ws + 4 * SD;
  float* t  = ws + 5 * SD;
  float* hm = ws + 6 * SD;                         // [S,F]
  float* concat = q;                               // alias: only live pre-layers (5.5M < 10.5M)

  dim3 blk256(256), blk128(128);
  dim3 gGather((S_LEN * CONCAT + 255) / 256);
  dim3 gAttn(S_LEN / 64, NHEAD);
  dim3 gLN(S_LEN);
  #define GEMM_GRID(Nv) dim3(((Nv) + 63) / 64, S_LEN / 128)

  // Embedding: gather -> GEMM(in_W)+in_b -> LN( +pos +tok )
  gather_kernel<<<gGather, blk256, 0, stream>>>(data, e0, e1, e2, e3, e4, e5, concat);
  gemm_kernel<<<GEMM_GRID(D_MODEL), blk256, 0, stream>>>(concat, in_W, in_b, t,
                                                         S_LEN, D_MODEL, CONCAT, 0);
  ln_kernel<<<gLN, blk256, 0, stream>>>(t, pos, tok, eg, eb, x);

  for (int l = 0; l < NLAYER; ++l) {
    const size_t wDD = (size_t)l * D_MODEL * D_MODEL;
    const size_t wDF = (size_t)l * D_MODEL * FFN;
    gemm_kernel<<<GEMM_GRID(D_MODEL), blk256, 0, stream>>>(x, Wq + wDD, bq + (size_t)l * D_MODEL, q,
                                                           S_LEN, D_MODEL, D_MODEL, 1);
    gemm_kernel<<<GEMM_GRID(D_MODEL), blk256, 0, stream>>>(x, Wk + wDD, bk + (size_t)l * D_MODEL, k,
                                                           S_LEN, D_MODEL, D_MODEL, 0);
    gemm_kernel<<<GEMM_GRID(D_MODEL), blk256, 0, stream>>>(x, Wv + wDD, bv + (size_t)l * D_MODEL, v,
                                                           S_LEN, D_MODEL, D_MODEL, 0);
    attn_kernel<<<gAttn, blk128, 0, stream>>>(q, k, v, masks, a);
    gemm_kernel<<<GEMM_GRID(D_MODEL), blk256, 0, stream>>>(a, Wo + wDD, bo + (size_t)l * D_MODEL, t,
                                                           S_LEN, D_MODEL, D_MODEL, 0);
    ln_kernel<<<gLN, blk256, 0, stream>>>(x, t, nullptr,
                                          ln1g + (size_t)l * D_MODEL, ln1b + (size_t)l * D_MODEL, x);
    gemm_kernel<<<GEMM_GRID(FFN), blk256, 0, stream>>>(x, Wf1 + wDF, bf1 + (size_t)l * FFN, hm,
                                                       S_LEN, FFN, D_MODEL, 2);
    gemm_kernel<<<GEMM_GRID(D_MODEL), blk256, 0, stream>>>(hm, Wf2 + wDF, bf2 + (size_t)l * D_MODEL, t,
                                                           S_LEN, D_MODEL, FFN, 0);
    ln_kernel<<<gLN, blk256, 0, stream>>>(x, t, nullptr,
                                          ln2g + (size_t)l * D_MODEL, ln2b + (size_t)l * D_MODEL, x);
  }

  gemm_kernel<<<GEMM_GRID(HEADV), blk256, 0, stream>>>(x, projW, projb, out,
                                                       S_LEN, HEADV, D_MODEL, 0);
  #undef GEMM_GRID
}
